// GCNRecommendationModel_3109556322453
// MI455X (gfx1250) — matrix operand order repaired and checked by reading, not compiled, not run on hardware
//
#include <hip/hip_runtime.h>

// GCN 2-layer: h1 = relu(GCNagg(x@W1)+b1); out = GCNagg(h1@W2)+b2
// N=100000 nodes, E=1.6M edges, DIN=256, HID=DOUT=64.

typedef __attribute__((ext_vector_type(16))) __bf16 v16bf;
typedef __attribute__((ext_vector_type(8)))  float  v8f;

#define DCOLS 64   // both layer outputs have 64 columns

// ---------------- degree / symmetric normalization ----------------
__global__ void k_init_deg(float* deg, int n) {
  int i = blockIdx.x * blockDim.x + threadIdx.x;
  if (i < n) deg[i] = 1.0f;                 // self-loop weight
}

__global__ void k_deg_scatter(const int* __restrict__ dst,
                              const float* __restrict__ ew,
                              float* deg, int e_cnt) {
  int e = blockIdx.x * blockDim.x + threadIdx.x;
  if (e < e_cnt) atomicAdd(&deg[dst[e]], ew[e]);
}

__global__ void k_make_dinv(float* deg, int n) {    // in-place deg -> deg^{-1/2}
  int i = blockIdx.x * blockDim.x + threadIdx.x;
  if (i < n) {
    float d = deg[i];
    deg[i] = (d > 0.0f) ? rsqrtf(d) : 0.0f;
  }
}

// ---------------- pack W into bf16 WMMA B-fragment order ----------------
// Fragment f = (kb/32)*4 + coltile. Lane's 16 bf16 elements stored contiguously
// (32 B, 32 B aligned) so the GEMM loads each fragment with 2x b128.
// B-fragment order used here (32x16, contiguous k, same as A): lanes 0-15 hold K=kb+0..15 (elem j <-> K=kb+j),
// lanes 16-31 hold K=kb+16..31; column = coltile*16 + (lane&15).
__global__ void k_pack_w(const float* __restrict__ W, __bf16* __restrict__ Wf, int K) {
  int t = blockIdx.x * blockDim.x + threadIdx.x;   // one thread per (fragment, lane)
  int nfrag = (K >> 5) * 4;
  if (t >= nfrag * 32) return;
  int f    = t >> 5;
  int lane = t & 31;
  int half = lane >> 4;
  int m    = lane & 15;
  int kB   = (f >> 2) * 32 + half * 16;
  int col  = (f & 3) * 16 + m;
  __bf16* o = Wf + (size_t)t * 16;
  #pragma unroll
  for (int j = 0; j < 16; ++j)
    o[j] = (__bf16)W[(size_t)(kB + j) * DCOLS + col];
}

// ---------------- WMMA GEMM: C[M x 64] = A[M x K] @ B[K x 64] ----------------
// One wave per 16-row tile; 4 accumulators cover all 64 output columns, so A
// is loaded exactly once per row (fp32, converted in-register). B comes from
// the pre-swizzled bf16 fragment buffer via vector loads.
__global__ void k_gemm_bf16(const float* __restrict__ A,
                            const __bf16* __restrict__ Bf,
                            float* __restrict__ C, int M, int K) {
  int wave = (blockIdx.x * blockDim.x + threadIdx.x) >> 5;
  int lane = threadIdx.x & 31;
  int row0 = wave * 16;
  if (row0 >= M) return;

  int half = lane >> 4;      // lane group 0: lanes 0-15, 1: lanes 16-31
  int m    = lane & 15;      // A row within tile / C column

  v8f acc[4];
  #pragma unroll
  for (int c = 0; c < 4; ++c) acc[c] = (v8f)0.0f;

  const float* arow = A + (size_t)(row0 + m) * K;

  for (int kb = 0; kb < K; kb += 32) {
    // A fragment, 16x32 bf16: lane-half g holds K = kb+g*16+{0..15}
    // in elements 0..15 (matches packed B fragment order).
    int ka = kb + half * 16;
    v16bf afrag;
    #pragma unroll
    for (int j = 0; j < 8; ++j) {
      afrag[j]     = (__bf16)arow[ka + j];
      afrag[j + 8] = (__bf16)arow[ka + 8 + j];
    }
    #pragma unroll
    for (int c = 0; c < 4; ++c) {
      v16bf bfrag = *(const v16bf*)(
          Bf + ((size_t)(((kb >> 5) * 4 + c) * 32 + lane)) * 16);
      acc[c] = __builtin_amdgcn_wmma_f32_16x16x32_bf16(
          /*neg_a=*/false, afrag, /*neg_b=*/false, bfrag,
          /*c_mod=*/(short)0, acc[c], /*reuse_a=*/false, /*reuse_b=*/false);
    }
  }

  // C/D 16x16 f32 layout: VGPR r -> row half*8+r, column = lane&15.
  #pragma unroll
  for (int c = 0; c < 4; ++c) {
    #pragma unroll
    for (int r = 0; r < 8; ++r)
      C[(size_t)(row0 + half * 8 + r) * DCOLS + c * 16 + m] = acc[c][r];
  }
}

// ---------------- self-loop init / epilogues (float2 vectorized) ----------------
// out[i] = h[i] * dinv[i]^2          (self-loop message, norm = dinv^2, ew=1)
__global__ void k_self_mul(const float* __restrict__ h,
                           const float* __restrict__ dinv,
                           float* __restrict__ out, int n) {
  int idx = blockIdx.x * blockDim.x + threadIdx.x;   // n*32 threads
  if (idx >= n * 32) return;
  int node = idx >> 5;
  int c    = (idx & 31) * 2;
  float s = dinv[node]; s *= s;
  float2 v = *(const float2*)(h + (size_t)node * DCOLS + c);
  float2 o; o.x = v.x * s; o.y = v.y * s;
  *(float2*)(out + (size_t)node * DCOLS + c) = o;
}

// out[i] = h[i] * dinv[i]^2 + bias   (layer-2 epilogue writes d_out; scatter adds later)
__global__ void k_self_mul_bias(const float* __restrict__ h,
                                const float* __restrict__ dinv,
                                const float* __restrict__ bias,
                                float* __restrict__ out, int n) {
  int idx = blockIdx.x * blockDim.x + threadIdx.x;
  if (idx >= n * 32) return;
  int node = idx >> 5;
  int c    = (idx & 31) * 2;
  float s = dinv[node]; s *= s;
  float2 v = *(const float2*)(h + (size_t)node * DCOLS + c);
  float2 o; o.x = v.x * s + bias[c]; o.y = v.y * s + bias[c + 1];
  *(float2*)(out + (size_t)node * DCOLS + c) = o;
}

// out[i] = relu(in[i] + bias)
__global__ void k_relu_bias(const float* __restrict__ in,
                            const float* __restrict__ bias,
                            float* __restrict__ out, int n) {
  int idx = blockIdx.x * blockDim.x + threadIdx.x;
  if (idx >= n * 32) return;
  int node = idx >> 5;
  int c    = (idx & 31) * 2;
  float2 v = *(const float2*)(in + (size_t)node * DCOLS + c);
  float2 o;
  o.x = fmaxf(v.x + bias[c], 0.0f);
  o.y = fmaxf(v.y + bias[c + 1], 0.0f);
  *(float2*)(out + (size_t)node * DCOLS + c) = o;
}

// ---------------- edge message scatter: one wave per edge ----------------
// out[dst] += h[src] * dinv[src]*ew*dinv[dst]   (64 channels, float2 per lane)
__global__ void k_edge_scatter(const int* __restrict__ src,
                               const int* __restrict__ dst,
                               const float* __restrict__ ew,
                               const float* __restrict__ dinv,
                               const float* __restrict__ h,
                               float* __restrict__ out, int e_cnt) {
  int e = (blockIdx.x * blockDim.x + threadIdx.x) >> 5;
  if (e >= e_cnt) return;
  int lane = threadIdx.x & 31;
  int s = src[e];
  int d = dst[e];
  float norm = dinv[s] * ew[e] * dinv[d];
  float2 v = *(const float2*)(h + (size_t)s * DCOLS + 2 * lane);
  float* o = out + (size_t)d * DCOLS + 2 * lane;
  atomicAdd(o,     v.x * norm);
  atomicAdd(o + 1, v.y * norm);
}

// ---------------- launch ----------------
extern "C" void kernel_launch(void* const* d_in, const int* in_sizes, int n_in,
                              void* d_out, int out_size, void* d_ws, size_t ws_size,
                              hipStream_t stream) {
  const float* x  = (const float*)d_in[0];     // [N, 256]
  const int*   ei = (const int*)  d_in[1];     // [2, E]
  const float* ew = (const float*)d_in[2];     // [E]
  const float* W1 = (const float*)d_in[3];     // [256, 64]
  const float* b1 = (const float*)d_in[4];     // [64]
  const float* W2 = (const float*)d_in[5];     // [64, 64]
  const float* b2 = (const float*)d_in[6];     // [64]
  float* out = (float*)d_out;                  // [N, 64]

  const int N   = in_sizes[0] / 256;           // 100000
  const int E   = in_sizes[2];                 // 1600000
  const int DIN = 256;

  const int* src = ei;
  const int* dst = ei + E;

  // Workspace (floats): dinv[N] | buf1[N*64] | buf2[N*64] | wf1,wf2 (bf16 frags)
  float* dinv = (float*)d_ws;
  float* buf1 = dinv + N;
  float* buf2 = buf1 + (size_t)N * DCOLS;
  __bf16* wf1 = (__bf16*)(buf2 + (size_t)N * DCOLS);  // 256*64 bf16 (32-B aligned)
  __bf16* wf2 = wf1 + (size_t)DIN * DCOLS;            // 64*64 bf16

  const int TPB = 256;
  int gN   = (N + TPB - 1) / TPB;              // node-parallel
  int gE   = (E + TPB - 1) / TPB;              // edge-parallel (1 thread/edge)
  int gEw  = (E + 7) / 8;                      // edge-parallel (1 wave/edge)
  int gNC  = (N * 32 + TPB - 1) / TPB;         // node*channel/2 parallel
  int gGem = ((N / 16) * 32 + TPB - 1) / TPB;  // 1 wave per 16-row tile
  int gP1  = ((DIN  / 32) * 4 * 32 + TPB - 1) / TPB;   // W1 frag-pack threads
  int gP2  = ((DCOLS / 32) * 4 * 32 + TPB - 1) / TPB;  // W2 frag-pack threads

  // normalization: deg = 1 (self loop) + sum(ew at dst); dinv = deg^{-1/2}
  k_init_deg   <<<gN,  TPB, 0, stream>>>(dinv, N);
  k_deg_scatter<<<gE,  TPB, 0, stream>>>(dst, ew, dinv, E);
  k_make_dinv  <<<gN,  TPB, 0, stream>>>(dinv, N);

  // pre-swizzle weights into WMMA B-fragment bf16 layout (one shot, tiny)
  k_pack_w<<<gP1, TPB, 0, stream>>>(W1, wf1, DIN);
  k_pack_w<<<gP2, TPB, 0, stream>>>(W2, wf2, DCOLS);

  // layer 1: h1 = x @ W1 ; agg = selfloop + edge scatter ; a1 = relu(agg + b1)
  k_gemm_bf16    <<<gGem, TPB, 0, stream>>>(x, wf1, buf1, N, DIN);
  k_self_mul     <<<gNC,  TPB, 0, stream>>>(buf1, dinv, buf2, N);
  k_edge_scatter <<<gEw,  TPB, 0, stream>>>(src, dst, ew, dinv, buf1, buf2, E);
  k_relu_bias    <<<gNC,  TPB, 0, stream>>>(buf2, b1, buf1, N);

  // layer 2: h2 = a1 @ W2 ; out = selfloop + b2 + edge scatter
  k_gemm_bf16    <<<gGem, TPB, 0, stream>>>(buf1, wf2, buf2, N, DCOLS);
  k_self_mul_bias<<<gNC,  TPB, 0, stream>>>(buf2, dinv, b2, out, N);
  k_edge_scatter <<<gEw,  TPB, 0, stream>>>(src, dst, ew, dinv, buf2, out, E);
}
